// CharacterBertCharacterCNN_38886633898184
// MI455X (gfx1250) — matrix operand order repaired and checked by reading, not compiled, not run on hardware
//
#include <hip/hip_runtime.h>
#include <hip/hip_bf16.h>
#include <stdint.h>

// ---- problem dims ----
#define N_TOK   4096      // 8*512 tokens
#define MAXCH   50
#define EMB     16
#define XSTRIDE 1152      // padded per-token bf16 row (covers K/M tile overreach, zero-filled)
#define TF      2048      // total filters
#define HWC     4096      // highway combined output cols
#define HID     768

typedef __attribute__((ext_vector_type(16))) __bf16 bf16x16;
typedef __attribute__((ext_vector_type(8)))  float  f32x8;

union FragAB { bf16x16 v; uint4 q[2]; };

__device__ __forceinline__ unsigned short f2bf(float f) {
  unsigned int x = __float_as_uint(f);
  x += 0x7FFFu + ((x >> 16) & 1u);          // round-to-nearest-even
  return (unsigned short)(x >> 16);
}

#define WMMA_BF16(a, b, c) \
  __builtin_amdgcn_wmma_f32_16x16x32_bf16(false, (a), false, (b), (short)0, (c), false, false)

// per-width tables (w, nf, col-tile start, K-steps of 32, M-tiles of 16, wconv elem offset)
__constant__ int c_w[7]      = {1,2,3,4,5,6,7};
__constant__ int c_nf[7]     = {32,32,64,128,256,512,1024};
__constant__ int c_tstart[7] = {0,2,4,8,16,32,64};
__constant__ int c_ksteps[7] = {1,1,2,2,3,3,4};
__constant__ int c_mtiles[7] = {4,4,3,3,3,3,3};
__constant__ int c_woff[7]   = {0,1024,2048,6144,14336,38912,88064};

// ---------------- weight packing: dense [K][C] f32 -> PB[(s*C+col)*32+kk] bf16 ----------------
__global__ void pack_dense_kernel(const float* __restrict__ W, unsigned short* __restrict__ PB,
                                  int K, int C) {
  size_t idx = (size_t)blockIdx.x * blockDim.x + threadIdx.x;
  size_t total = (size_t)K * C;             // K multiple of 32 here
  if (idx >= total) return;
  int cs = C * 32;
  int s   = (int)(idx / cs);
  int rem = (int)(idx % cs);
  int col = rem >> 5;
  int kk  = rem & 31;
  int k = s * 32 + kk;
  PB[idx] = f2bf(W[(size_t)k * C + col]);
}

// conv weights (nf,16,w) -> PB for B[k=dx*16+c][f], K padded to 32*ksteps with zeros
__global__ void pack_conv_kernel(const float* __restrict__ Wf, unsigned short* __restrict__ PB,
                                 int nf, int w, int ksteps) {
  int total = ksteps * nf * 32;
  int idx = blockIdx.x * blockDim.x + threadIdx.x;
  if (idx >= total) return;
  int fs  = nf * 32;
  int s   = idx / fs;
  int rem = idx % fs;
  int f   = rem >> 5;
  int kk  = rem & 31;
  int k = s * 32 + kk;
  float v = 0.0f;
  if (k < 16 * w) {
    int c = k & 15, dx = k >> 4;
    v = Wf[(size_t)f * 16 * w + c * w + dx];
  }
  PB[idx] = f2bf(v);
}

// ---------------- embedding gather -> bf16 x[n,1152] (zero padded) ----------------
__global__ void embed_kernel(const int* __restrict__ ids, const float* __restrict__ emb,
                             unsigned short* __restrict__ xemb) {
  size_t idx = (size_t)blockIdx.x * blockDim.x + threadIdx.x;
  if (idx >= (size_t)N_TOK * XSTRIDE) return;
  int n = (int)(idx / XSTRIDE);
  int q = (int)(idx % XSTRIDE);
  float v = 0.0f;
  if (q < MAXCH * EMB) {
    int p = q >> 4, c = q & 15;
    int id = ids[(size_t)n * MAXCH + p];
    v = emb[(size_t)id * EMB + c];
  }
  xemb[idx] = f2bf(v);
}

// ---------------- conv + maxpool + relu via WMMA ----------------
// one wave per (token, 16-filter column tile); block (32,8)
__global__ void __launch_bounds__(256) conv_maxpool_kernel(
    const unsigned short* __restrict__ xemb,
    const unsigned short* __restrict__ wconv,
    const float* __restrict__ biascat,
    float* __restrict__ t0f, unsigned short* __restrict__ t0b) {
  const int l  = threadIdx.x;
  const int ct = blockIdx.x * 8 + threadIdx.y;   // 0..127 column tile over 2048 filters
  const int n  = blockIdx.y;
  int wi = 0;
  #pragma unroll
  for (int i = 1; i < 7; ++i) if (ct >= c_tstart[i]) wi = i;
  const int nf = c_nf[wi], ks = c_ksteps[wi], mt = c_mtiles[wi];
  const int valid = 51 - c_w[wi];
  const int lct = ct - c_tstart[wi];
  const unsigned short* wbase = wconv + c_woff[wi];
  const int lo16 = l & 15;
  const int hi   = l >> 4;
  const unsigned short* arow = xemb + (size_t)n * XSTRIDE;

  float m[8];
  #pragma unroll
  for (int r = 0; r < 8; ++r) m[r] = -3.0e38f;

  for (int mtile = 0; mtile < mt; ++mtile) {
    f32x8 acc = {};
    const int row = mtile * 16 + lo16;            // A row this lane holds
    for (int s = 0; s < ks; ++s) {
      FragAB a, b;
      const unsigned short* ap = arow + row * EMB + s * 32 + hi * 8;
      a.q[0] = *(const uint4*)(ap);               // K {0..7} / {8..15}
      a.q[1] = *(const uint4*)(ap + 16);          // K {16..23} / {24..31}
      const unsigned short* bp = wbase + ((size_t)(s * nf + lct * 16 + lo16)) * 32 + hi * 8;
      b.q[0] = *(const uint4*)(bp);
      b.q[1] = *(const uint4*)(bp + 16);
      acc = WMMA_BF16(a.v, b.v, acc);
    }
    #pragma unroll
    for (int r = 0; r < 8; ++r) {
      int orow = mtile * 16 + r + hi * 8;         // D layout: lanes>=16 hold M=r+8
      if (orow < valid) m[r] = fmaxf(m[r], acc[r]);
    }
  }
  float mm = m[0];
  #pragma unroll
  for (int r = 1; r < 8; ++r) mm = fmaxf(mm, m[r]);
  mm = fmaxf(mm, __shfl_xor(mm, 16, 32));         // combine the two M halves
  if (l < 16) {
    int gcol = ct * 16 + l;
    float v = fmaxf(mm + biascat[gcol], 0.0f);    // bias is position-invariant: max then +b then relu
    t0f[(size_t)n * TF + gcol] = v;
    t0b[(size_t)n * TF + gcol] = f2bf(v);
  }
}

// ---------------- highway layer: register-blocked dual-GEMM (nl + gate) + gating ----------------
// one wave computes a 64-row x 32-col tile (x2 for gate cols) => 16 WMMAs per K-step
#define HW_MT 4
#define HW_CT 2
__global__ void __launch_bounds__(32) highway_kernel(
    const unsigned short* __restrict__ tinb, const float* __restrict__ tinf,
    const unsigned short* __restrict__ pb,   // packed [64][4096][32]
    const float* __restrict__ hwb,           // [4096] bias
    float* __restrict__ toutf, unsigned short* __restrict__ toutb) {
  const int l  = threadIdx.x;
  const int nb = blockIdx.x;                     // 0..63 -> cols nb*32
  const int mb = blockIdx.y;                     // 0..63 -> rows mb*64
  const int lo16 = l & 15;
  const int hi   = l >> 4;
  const int row0 = mb * 64 + lo16;
  const int col0 = nb * 32 + lo16;

  f32x8 accN[HW_MT][HW_CT] = {};
  f32x8 accG[HW_MT][HW_CT] = {};

  const unsigned short* ap0 = tinb + (size_t)row0 * TF + hi * 8;
  const unsigned short* bp0 = pb + (size_t)col0 * 32 + hi * 8;

  for (int s = 0; s < 64; ++s) {
    FragAB a[HW_MT], bn[HW_CT], bg[HW_CT];
    #pragma unroll
    for (int i = 0; i < HW_MT; ++i) {
      const unsigned short* ap = ap0 + (size_t)i * 16 * TF + s * 32;
      a[i].q[0] = *(const uint4*)(ap);
      a[i].q[1] = *(const uint4*)(ap + 16);
    }
    #pragma unroll
    for (int j = 0; j < HW_CT; ++j) {
      const unsigned short* bnp = bp0 + (size_t)s * HWC * 32 + (size_t)j * 16 * 32;
      const unsigned short* bgp = bnp + (size_t)TF * 32;
      bn[j].q[0] = *(const uint4*)(bnp);
      bn[j].q[1] = *(const uint4*)(bnp + 16);
      bg[j].q[0] = *(const uint4*)(bgp);
      bg[j].q[1] = *(const uint4*)(bgp + 16);
    }
    __builtin_prefetch(bp0 + (size_t)(s + 1) * HWC * 32, 0, 1);  // next B K-slab
    #pragma unroll
    for (int i = 0; i < HW_MT; ++i) {
      #pragma unroll
      for (int j = 0; j < HW_CT; ++j) {
        accN[i][j] = WMMA_BF16(a[i].v, bn[j].v, accN[i][j]);
        accG[i][j] = WMMA_BF16(a[i].v, bg[j].v, accG[i][j]);
      }
    }
  }
  #pragma unroll
  for (int j = 0; j < HW_CT; ++j) {
    const int col = nb * 32 + j * 16 + lo16;
    const float bN = hwb[col];
    const float bG = hwb[col + TF];
    #pragma unroll
    for (int i = 0; i < HW_MT; ++i) {
      #pragma unroll
      for (int r = 0; r < 8; ++r) {
        int row = mb * 64 + i * 16 + r + hi * 8;
        float told = tinf[(size_t)row * TF + col];
        float nl = fmaxf(accN[i][j][r] + bN, 0.0f);
        float g  = 1.0f / (1.0f + __expf(-(accG[i][j][r] + bG)));
        float tn = g * told + (1.0f - g) * nl;
        toutf[(size_t)row * TF + col] = tn;
        toutb[(size_t)row * TF + col] = f2bf(tn);
      }
    }
  }
}

// ---------------- projection GEMM + bias -> f32 out ----------------
// one wave computes a 64x64 tile => 16 WMMAs per K-step
#define PJ_MT 4
#define PJ_CT 4
__global__ void __launch_bounds__(32) proj_kernel(
    const unsigned short* __restrict__ tinb,
    const unsigned short* __restrict__ pb,   // packed [64][768][32]
    const float* __restrict__ pbias,
    float* __restrict__ out) {
  const int l  = threadIdx.x;
  const int nb = blockIdx.x;                     // 0..11 -> cols nb*64
  const int mb = blockIdx.y;                     // 0..63 -> rows mb*64
  const int lo16 = l & 15;
  const int hi   = l >> 4;
  const int row0 = mb * 64 + lo16;
  const int col0 = nb * 64 + lo16;

  f32x8 acc[PJ_MT][PJ_CT] = {};

  const unsigned short* ap0 = tinb + (size_t)row0 * TF + hi * 8;
  const unsigned short* bp0 = pb + (size_t)col0 * 32 + hi * 8;

  for (int s = 0; s < 64; ++s) {
    FragAB a[PJ_MT], b[PJ_CT];
    #pragma unroll
    for (int i = 0; i < PJ_MT; ++i) {
      const unsigned short* ap = ap0 + (size_t)i * 16 * TF + s * 32;
      a[i].q[0] = *(const uint4*)(ap);
      a[i].q[1] = *(const uint4*)(ap + 16);
    }
    #pragma unroll
    for (int j = 0; j < PJ_CT; ++j) {
      const unsigned short* bp = bp0 + (size_t)s * HID * 32 + (size_t)j * 16 * 32;
      b[j].q[0] = *(const uint4*)(bp);
      b[j].q[1] = *(const uint4*)(bp + 16);
    }
    #pragma unroll
    for (int i = 0; i < PJ_MT; ++i) {
      #pragma unroll
      for (int j = 0; j < PJ_CT; ++j) {
        acc[i][j] = WMMA_BF16(a[i].v, b[j].v, acc[i][j]);
      }
    }
  }
  #pragma unroll
  for (int j = 0; j < PJ_CT; ++j) {
    const int col = nb * 64 + j * 16 + lo16;
    const float bb = pbias[col];
    #pragma unroll
    for (int i = 0; i < PJ_MT; ++i) {
      #pragma unroll
      for (int r = 0; r < 8; ++r) {
        int row = mb * 64 + i * 16 + r + hi * 8;
        out[(size_t)row * HID + col] = acc[i][j][r] + bb;
      }
    }
  }
}

// ---------------- launch ----------------
extern "C" void kernel_launch(void* const* d_in, const int* in_sizes, int n_in,
                              void* d_out, int out_size, void* d_ws, size_t ws_size,
                              hipStream_t stream) {
  const int*   ids  = (const int*)d_in[0];
  const float* cemb = (const float*)d_in[1];
  const float* convw[7]; const float* convb[7];
  for (int i = 0; i < 7; ++i) { convw[i] = (const float*)d_in[2 + 2*i]; convb[i] = (const float*)d_in[3 + 2*i]; }
  const float* hww0 = (const float*)d_in[16];
  const float* hwb0 = (const float*)d_in[17];
  const float* hww1 = (const float*)d_in[18];
  const float* hwb1 = (const float*)d_in[19];
  const float* pjw  = (const float*)d_in[20];
  const float* pjb  = (const float*)d_in[21];
  float* out = (float*)d_out;

  char* ws = (char*)d_ws;
  // workspace layout (all offsets multiples of 256)
  const size_t OFF_XEMB  = 0;                         // 4096*1152 bf16      = 9,437,184 B
  const size_t OFF_WCONV = OFF_XEMB  + 9437184;       // 219136 bf16         =   438,272 B
  const size_t OFF_WHW0  = OFF_WCONV + 438272;        // 2048*4096 bf16      = 16,777,216 B
  const size_t OFF_WHW1  = OFF_WHW0  + 16777216;
  const size_t OFF_WPROJ = OFF_WHW1  + 16777216;      // 2048*768 bf16       = 3,145,728 B
  const size_t OFF_BIASC = OFF_WPROJ + 3145728;       // 2048 f32
  const size_t OFF_T0F   = OFF_BIASC + 8192;          // 4096*2048 f32       = 33,554,432 B
  const size_t OFF_T0B   = OFF_T0F   + 33554432;      // 4096*2048 bf16
  const size_t OFF_T1F   = OFF_T0B   + 16777216;
  const size_t OFF_T1B   = OFF_T1F   + 33554432;      // end ~147 MB

  unsigned short* xemb  = (unsigned short*)(ws + OFF_XEMB);
  unsigned short* wconv = (unsigned short*)(ws + OFF_WCONV);
  unsigned short* whw0  = (unsigned short*)(ws + OFF_WHW0);
  unsigned short* whw1  = (unsigned short*)(ws + OFF_WHW1);
  unsigned short* wproj = (unsigned short*)(ws + OFF_WPROJ);
  float*          bcat  = (float*)(ws + OFF_BIASC);
  float*          t0f   = (float*)(ws + OFF_T0F);
  unsigned short* t0b   = (unsigned short*)(ws + OFF_T0B);
  float*          t1f   = (float*)(ws + OFF_T1F);
  unsigned short* t1b   = (unsigned short*)(ws + OFF_T1B);

  // 1) pack conv weights (per width)
  const int nf_h[7]     = {32,32,64,128,256,512,1024};
  const int w_h[7]      = {1,2,3,4,5,6,7};
  const int ksteps_h[7] = {1,1,2,2,3,3,4};
  const int woff_h[7]   = {0,1024,2048,6144,14336,38912,88064};
  for (int i = 0; i < 7; ++i) {
    int total = ksteps_h[i] * nf_h[i] * 32;
    pack_conv_kernel<<<(total + 255) / 256, 256, 0, stream>>>(
        convw[i], wconv + woff_h[i], nf_h[i], w_h[i], ksteps_h[i]);
  }
  // 2) pack dense weights
  pack_dense_kernel<<<(2048u*4096u)/256u, 256, 0, stream>>>(hww0, whw0, 2048, 4096);
  pack_dense_kernel<<<(2048u*4096u)/256u, 256, 0, stream>>>(hww1, whw1, 2048, 4096);
  pack_dense_kernel<<<(2048u*768u)/256u,  256, 0, stream>>>(pjw,  wproj, 2048, 768);
  // 3) concat conv biases
  const int fbase_h[7] = {0,32,64,128,256,512,1024};
  for (int i = 0; i < 7; ++i)
    hipMemcpyAsync(bcat + fbase_h[i], convb[i], (size_t)nf_h[i] * sizeof(float),
                   hipMemcpyDeviceToDevice, stream);
  // 4) embedding gather -> bf16 (zero-padded rows)
  embed_kernel<<<((size_t)N_TOK * XSTRIDE + 255) / 256, 256, 0, stream>>>(ids, cemb, xemb);
  // 5) conv + maxpool + relu
  conv_maxpool_kernel<<<dim3(16, N_TOK), dim3(32, 8), 0, stream>>>(xemb, wconv, bcat, t0f, t0b);
  // 6) highway x2 (ping-pong), 64x64 register-blocked tiles
  highway_kernel<<<dim3(64, 64), dim3(32), 0, stream>>>(t0b, t0f, whw0, hwb0, t1f, t1b);
  highway_kernel<<<dim3(64, 64), dim3(32), 0, stream>>>(t1b, t1f, whw1, hwb1, t0f, t0b);
  // 7) projection -> out, 64x64 register-blocked tiles
  proj_kernel<<<dim3(12, 64), dim3(32), 0, stream>>>(t0b, wproj, pjb, out);
  (void)in_sizes; (void)n_in; (void)out_size; (void)ws_size;
}
